// Seg3dLossless_16226386444604
// MI455X (gfx1250) — matrix operand order repaired and checked by reading, not compiled, not run on hardware
//
#include <hip/hip_runtime.h>
#include <hip/hip_bf16.h>
#include <math.h>

typedef __attribute__((ext_vector_type(16))) _Float16 v16h;
typedef __attribute__((ext_vector_type(8)))  float    v8f;

#define RF_   129
#define NFRAG 14   // 4 (L1) + 8 (L2) + 2 (L3) A-matrix (W^T) fragments

#define TILES_PER_WAVE 4
#define WAVES_PER_BLOCK 8

// ---------------------------------------------------------------------------
// Transposed formulation: H_{l+1} = relu(W_l^T @ H_l + b). Activations are the
// B operand (features = K down the halves, points across lanes); weights W^T
// are the A operand (output feature = row M across lanes).
//
// A-fragment packing used here (16-bit, 16x32, contiguous k): lane l holds row M = l&15;
//   half i -> K = i + 16*(l>=16)
// Fragment f, lane l occupies wpack[(f*32+l)*16 .. +16)
//
// Fragments: f=0..3   layer1 W1^T tiles (K: 0-2 = w1 rows, K=3 = b1 bias col)
//            f=4..11  layer2 W2^T tiles (nt = (f-4)>>1, kc = (f-4)&1)
//            f=12..13 layer3 W3^T (row 0 only)
// ---------------------------------------------------------------------------
__global__ void pack_weights_kernel(const float* __restrict__ w1,
                                    const float* __restrict__ b1,
                                    const float* __restrict__ w2,
                                    const float* __restrict__ w3,
                                    _Float16* __restrict__ wpack) {
  int t = blockIdx.x * blockDim.x + threadIdx.x;
  if (t >= NFRAG * 32) return;
  int f = t >> 5, l = t & 31;
  int M = l & 15;
  int hi = l >> 4;
  _Float16* dst = wpack + ((size_t)(f * 32 + l)) * 16;
  #pragma unroll
  for (int i = 0; i < 16; i++) {
    int K = i + 16 * hi;
    float v = 0.0f;
    if (f < 4) {                          // layer1: A[n][k] = w1[k][n], k=3 -> b1
      int nf = f * 16 + M;
      if (K < 3)       v = w1[K * 64 + nf];
      else if (K == 3) v = b1[nf];
    } else if (f < 12) {                  // layer2: A[n][k] = w2[k][n]
      int nt = (f - 4) >> 1, kc = (f - 4) & 1;
      int nf = nt * 16 + M;
      v = w2[(kc * 32 + K) * 64 + nf];
    } else {                              // layer3: row 0 = w3
      int kc = f - 12;
      if (M == 0) v = w3[kc * 32 + K];
    }
    dst[i] = (_Float16)v;
  }
}

static __device__ __forceinline__ unsigned int pack2h(float a, float b) {
  union { _Float16 h[2]; unsigned int u; } t;
  t.h[0] = (_Float16)a; t.h[1] = (_Float16)b;
  return t.u;
}

// Build B fragment (one K-chunk) for the next layer from packed activations.
// He = feature tile 2*kc (features kc*32 + 0..15), Ho = tile 2*kc+1.
// Lane l and lane l^16 hold complementary feature groups of the same point:
// exchange with ds_swizzle SWAPX16 (offset 0x401F). No dynamic indexing:
// pure value selects on hi (-> v_cndmask).
static __device__ __forceinline__ v16h build_bfrag(const unsigned int* He,
                                                   const unsigned int* Ho,
                                                   bool hi) {
  union { v16h h; unsigned int u[8]; } r;
  #pragma unroll
  for (int d = 0; d < 4; d++) {
    unsigned int znd = hi ? He[d] : Ho[d];  // what the partner lane needs
    unsigned int rec = (unsigned int)__builtin_amdgcn_ds_swizzle((int)znd, 0x401F);
    r.u[d]     = hi ? rec   : He[d];  // halves 0-7:  features kc*32 + 16*hi + [0..7]
    r.u[4 + d] = hi ? Ho[d] : rec;    // halves 8-15: features kc*32 + 16*hi + [8..15]
  }
  return r.h;
}

// ---------------------------------------------------------------------------
// WMMA MLP: one wave evaluates TILES_PER_WAVE tiles of 16 grid points; all
// weight fragments and biases loaded once per wave, held in VGPRs. No LDS,
// no barriers. R,S compile-time so div/mod strength-reduce.
// ---------------------------------------------------------------------------
template <int R, int S>
__global__ __launch_bounds__(32 * WAVES_PER_BLOCK) void eval_mlp_kernel(
    const _Float16* __restrict__ wpack,
    const float* __restrict__ b2g, const float* __restrict__ b3g,
    float* __restrict__ out) {
  constexpr int NPTS   = R * R * R;
  constexpr int NTILES = (NPTS + 15) / 16;

  int lane = threadIdx.x & 31;
  int wv   = threadIdx.x >> 5;
  int p    = lane & 15;          // point within tile
  bool hi  = (lane >> 4) != 0;
  int tile0 = (blockIdx.x * WAVES_PER_BLOCK + wv) * TILES_PER_WAVE;

  // ---- hoisted: weight fragments (A operands) ----
  const v16h* wp = (const v16h*)wpack;
  v16h A1[4], A2[8], A3[2];
  #pragma unroll
  for (int nt = 0; nt < 4; nt++) A1[nt] = wp[nt * 32 + lane];
  #pragma unroll
  for (int q = 0; q < 8; q++)    A2[q]  = wp[(4 + q) * 32 + lane];
  #pragma unroll
  for (int kc = 0; kc < 2; kc++) A3[kc] = wp[(12 + kc) * 32 + lane];

  // ---- hoisted: layer2 biases for this lane's accumulator rows ----
  int boff = hi ? 8 : 0;
  float bias2[4][8];
  #pragma unroll
  for (int nt = 0; nt < 4; nt++) {
    #pragma unroll
    for (int j = 0; j < 8; j++) bias2[nt][j] = b2g[nt * 16 + boff + j];
  }
  float b3 = b3g[0];

  const float scale = 2.0f / (float)RF_;

  #pragma unroll 1
  for (int tt = 0; tt < TILES_PER_WAVE; tt++) {
    int tile = tile0 + tt;
    if (tile >= NTILES) break;            // wave-uniform
    int g0 = tile * 16 + p;               // may exceed NPTS-1 in last tile:
    int ix = g0 % R;                      // harmless garbage, store is guarded
    int iy = (g0 / R) % R;
    int iz = g0 / (R * R);
    float fx = ((float)(ix * S) + 0.5f) * scale - 1.0f;
    float fy = ((float)(iy * S) + 0.5f) * scale - 1.0f;
    float fz = ((float)(iz * S) + 0.5f) * scale - 1.0f;

    // ---- B operand for layer1 (branch-free): K=0..2 coords, K=3 = 1.0 ----
    union { v16h h; unsigned int u[8]; } bb;
    unsigned int c01 = pack2h(fx, fy);
    unsigned int c23 = pack2h(fz, 1.0f);
    bb.u[0] = hi ? 0u : c01;
    bb.u[1] = hi ? 0u : c23;
    #pragma unroll
    for (int d = 2; d < 8; d++) bb.u[d] = 0u;
    v16h bf = bb.h;

    // ---- Layer 1: H1 = relu(W1^T P + b1)  (bias folded into K=3) ----
    unsigned int Hw[4][4];
    #pragma unroll
    for (int nt = 0; nt < 4; nt++) {
      v8f c = {};
      c = __builtin_amdgcn_wmma_f32_16x16x32_f16(false, A1[nt], false, bf,
                                                 (short)0, c, false, false);
      #pragma unroll
      for (int d = 0; d < 4; d++) {
        float h0 = c[2 * d],     h1 = c[2 * d + 1];
        h0 = h0 > 0.0f ? h0 : 0.0f;
        h1 = h1 > 0.0f ? h1 : 0.0f;
        Hw[nt][d] = pack2h(h0, h1);
      }
    }

    // ---- Layer 2: H2 = relu(W2^T H1 + b2) ----
    v16h B0 = build_bfrag(Hw[0], Hw[1], hi);
    v16h B1 = build_bfrag(Hw[2], Hw[3], hi);
    unsigned int Hw2[4][4];
    #pragma unroll
    for (int nt = 0; nt < 4; nt++) {
      v8f c = {};
      c = __builtin_amdgcn_wmma_f32_16x16x32_f16(false, A2[nt * 2 + 0], false, B0,
                                                 (short)0, c, false, false);
      c = __builtin_amdgcn_wmma_f32_16x16x32_f16(false, A2[nt * 2 + 1], false, B1,
                                                 (short)0, c, false, false);
      #pragma unroll
      for (int d = 0; d < 4; d++) {
        float h0 = c[2 * d]     + bias2[nt][2 * d];
        float h1 = c[2 * d + 1] + bias2[nt][2 * d + 1];
        h0 = h0 > 0.0f ? h0 : 0.0f;
        h1 = h1 > 0.0f ? h1 : 0.0f;
        Hw2[nt][d] = pack2h(h0, h1);
      }
    }

    // ---- Layer 3: z = W3^T H2 + b3 (row 0 -> lanes 0-15, VGPR 0) ----
    v16h C0 = build_bfrag(Hw2[0], Hw2[1], hi);
    v16h C1 = build_bfrag(Hw2[2], Hw2[3], hi);
    v8f c3 = {};
    c3 = __builtin_amdgcn_wmma_f32_16x16x32_f16(false, A3[0], false, C0,
                                                (short)0, c3, false, false);
    c3 = __builtin_amdgcn_wmma_f32_16x16x32_f16(false, A3[1], false, C1,
                                                (short)0, c3, false, false);

    if (!hi && g0 < NPTS) {
      float z = c3[0] + b3;
      out[g0] = 1.0f / (1.0f + __expf(-z));   // coalesced 16-lane store
    }
  }
}

// ---------------------------------------------------------------------------
// calculated[RF^3] = 1 at stride-8 lattice (level-17 points), else 0
// ---------------------------------------------------------------------------
__global__ void init_calc_kernel(unsigned char* __restrict__ calc) {
  int idx = blockIdx.x * blockDim.x + threadIdx.x;
  constexpr int n = RF_ * RF_ * RF_;
  if (idx >= n) return;
  int x = idx % RF_, y = (idx / RF_) % RF_, z = idx / (RF_ * RF_);
  calc[idx] = ((x % 8) == 0 && (y % 8) == 0 && (z % 8) == 0) ? 1 : 0;
}

// ---------------------------------------------------------------------------
// Trilinear upsample x2: src Rp^3 -> occ_i R^3; b0 = interp(src>0.5) in (0,1)
// ---------------------------------------------------------------------------
template <int R, int Rp>
__global__ void resize_kernel(const float* __restrict__ src,
                              float* __restrict__ occ_i,
                              unsigned char* __restrict__ b0) {
  int idx = blockIdx.x * blockDim.x + threadIdx.x;
  constexpr int n = R * R * R;
  if (idx >= n) return;
  int ox = idx % R, oy = (idx / R) % R, oz = idx / (R * R);
  int x0 = ox >> 1, y0 = oy >> 1, z0 = oz >> 1;
  float tx = (ox & 1) ? 0.5f : 0.0f;
  float ty = (oy & 1) ? 0.5f : 0.0f;
  float tz = (oz & 1) ? 0.5f : 0.0f;
  int x1 = x0 + 1 < Rp ? x0 + 1 : Rp - 1;
  int y1 = y0 + 1 < Rp ? y0 + 1 : Rp - 1;
  int z1 = z0 + 1 < Rp ? z0 + 1 : Rp - 1;
  float occ = 0.0f, val = 0.0f;
  #pragma unroll
  for (int dz = 0; dz < 2; dz++) {
    int z = dz ? z1 : z0; float wz = dz ? tz : (1.0f - tz);
    #pragma unroll
    for (int dy = 0; dy < 2; dy++) {
      int y = dy ? y1 : y0; float wy = dy ? ty : (1.0f - ty);
      #pragma unroll
      for (int dx = 0; dx < 2; dx++) {
        int x = dx ? x1 : x0; float wx = dx ? tx : (1.0f - tx);
        float w = wz * wy * wx;
        float v = src[((size_t)z * Rp + y) * Rp + x];
        occ += w * v;
        val += w * (v > 0.5f ? 1.0f : 0.0f);
      }
    }
  }
  occ_i[idx] = occ;
  b0[idx] = (val > 0.0f && val < 1.0f) ? 1 : 0;
}

// ---------------------------------------------------------------------------
// out = dilate3x3x3(m) & ~calculated[::s]   (boundary and nbr)
// ---------------------------------------------------------------------------
template <int R, int S>
__global__ void dilate_kernel(const unsigned char* __restrict__ m,
                              unsigned char* __restrict__ out,
                              const unsigned char* __restrict__ calc) {
  int idx = blockIdx.x * blockDim.x + threadIdx.x;
  constexpr int n = R * R * R;
  if (idx >= n) return;
  int ox = idx % R, oy = (idx / R) % R, oz = idx / (R * R);
  unsigned char v = 0;
  for (int dz = -1; dz <= 1; dz++) {
    int z = oz + dz; if (z < 0 || z >= R) continue;
    for (int dy = -1; dy <= 1; dy++) {
      int y = oy + dy; if (y < 0 || y >= R) continue;
      for (int dx = -1; dx <= 1; dx++) {
        int x = ox + dx; if (x < 0 || x >= R) continue;
        v |= m[((size_t)z * R + y) * R + x];
      }
    }
  }
  size_t cidx = ((size_t)(oz * S) * RF_ + (size_t)(oy * S)) * RF_ + (size_t)(ox * S);
  out[idx] = (v && !calc[cidx]) ? 1 : 0;
}

// ---------------------------------------------------------------------------
// occ = mask ? occ_true : occ (in place); conflict = mask & signflip;
// calculated[::s] |= mask.   (Serves both the boundary step and the conflict
// iterations: occ holds occ_i before the first call.)
// ---------------------------------------------------------------------------
template <int R, int S>
__global__ void fuse_kernel(float* __restrict__ occ,
                            const float* __restrict__ occ_true,
                            const unsigned char* __restrict__ mask,
                            unsigned char* __restrict__ conflict,
                            unsigned char* __restrict__ calc) {
  int idx = blockIdx.x * blockDim.x + threadIdx.x;
  constexpr int n = R * R * R;
  if (idx >= n) return;
  int ox = idx % R, oy = (idx / R) % R, oz = idx / (R * R);
  unsigned char mk = mask[idx];
  float t = occ_true[idx];
  float prev = occ[idx];
  if (mk) {
    conflict[idx] = ((prev - 0.5f) * (t - 0.5f) < 0.0f) ? 1 : 0;
    occ[idx] = t;
    calc[((size_t)(oz * S) * RF_ + (size_t)(oy * S)) * RF_ + (size_t)(ox * S)] = 1;
  } else {
    conflict[idx] = 0;
  }
}

// ---------------------------------------------------------------------------
template <int R, int S>
static void launch_eval_t(float* dst, const _Float16* wpack,
                          const float* b2, const float* b3, hipStream_t stream) {
  constexpr int npts = R * R * R;
  constexpr int ntiles = (npts + 15) / 16;
  int tpb = TILES_PER_WAVE * WAVES_PER_BLOCK;
  int blocks = (ntiles + tpb - 1) / tpb;
  eval_mlp_kernel<R, S><<<blocks, 32 * WAVES_PER_BLOCK, 0, stream>>>(
      wpack, b2, b3, dst);
}

template <int R, int S, int Rp>
static void run_level_t(const float* occ_prev, float* occ, float* bufT,
                        unsigned char* calc, unsigned char* mb0,
                        unsigned char* mbd, unsigned char* mcf,
                        const _Float16* wpack,
                        const float* b2, const float* b3, hipStream_t stream) {
  constexpr int n = R * R * R;
  int tb = 256, gb = (n + tb - 1) / tb;
  resize_kernel<R, Rp><<<gb, tb, 0, stream>>>(occ_prev, occ, mb0);
  dilate_kernel<R, S><<<gb, tb, 0, stream>>>(mb0, mbd, calc);    // boundary
  launch_eval_t<R, S>(bufT, wpack, b2, b3, stream);              // occ_true
  fuse_kernel<R, S><<<gb, tb, 0, stream>>>(occ, bufT, mbd, mcf, calc);
  for (int it = 0; it < 2; ++it) {
    dilate_kernel<R, S><<<gb, tb, 0, stream>>>(mcf, mb0, calc);  // nbr -> mb0
    fuse_kernel<R, S><<<gb, tb, 0, stream>>>(occ, bufT, mb0, mcf, calc);
  }
}

extern "C" void kernel_launch(void* const* d_in, const int* in_sizes, int n_in,
                              void* d_out, int out_size, void* d_ws, size_t ws_size,
                              hipStream_t stream) {
  (void)in_sizes; (void)n_in; (void)out_size; (void)ws_size;
  const float* w1 = (const float*)d_in[0];
  const float* b1 = (const float*)d_in[1];
  const float* w2 = (const float*)d_in[2];
  const float* b2 = (const float*)d_in[3];
  const float* w3 = (const float*)d_in[4];
  const float* b3 = (const float*)d_in[5];
  float* out = (float*)d_out;

  const size_t N129 = (size_t)RF_ * RF_ * RF_;
  char* ws = (char*)d_ws;
  size_t off = 0;
  _Float16* wpack = (_Float16*)(ws + off); off += ((size_t)NFRAG * 512 * 2 + 255) / 256 * 256;
  float* bufA = (float*)(ws + off); off += N129 * sizeof(float);
  float* bufB = (float*)(ws + off); off += N129 * sizeof(float);
  float* bufT = (float*)(ws + off); off += N129 * sizeof(float);
  unsigned char* calc = (unsigned char*)(ws + off); off += (N129 + 255) / 256 * 256;
  unsigned char* mb0  = (unsigned char*)(ws + off); off += (N129 + 255) / 256 * 256;
  unsigned char* mbd  = (unsigned char*)(ws + off); off += (N129 + 255) / 256 * 256;
  unsigned char* mcf  = (unsigned char*)(ws + off); off += (N129 + 255) / 256 * 256;

  pack_weights_kernel<<<(NFRAG * 32 + 255) / 256, 256, 0, stream>>>(w1, b1, w2, w3, wpack);

  {
    int tb = 256, gb = (int)((N129 + tb - 1) / tb);
    init_calc_kernel<<<gb, tb, 0, stream>>>(calc);
  }

  // Level 17 (base): full MLP evaluation
  launch_eval_t<17, 8>(bufA, wpack, b2, b3, stream);

  // Refinement levels 33 -> 65 -> 129 (final occ written directly to d_out)
  run_level_t<33, 4, 17>(bufA, bufB, bufT, calc, mb0, mbd, mcf, wpack, b2, b3, stream);
  run_level_t<65, 2, 33>(bufB, bufA, bufT, calc, mb0, mbd, mcf, wpack, b2, b3, stream);
  run_level_t<129, 1, 65>(bufA, out,  bufT, calc, mb0, mbd, mcf, wpack, b2, b3, stream);
}
